// NEmonForwardStep_84782654423308
// MI455X (gfx1250) — matrix operand order repaired and checked by reading, not compiled, not run on hardware
//
#include <hip/hip_runtime.h>
#include <hip/hip_bf16.h>

// ---------------------------------------------------------------------------
// monDEQ forward for MI455X (gfx1250, wave32, WMMA + async tensor staging).
// GEMMs run on v_wmma_f32_16x16x32_bf16 with an error-compensated hi/lo bf16
// split (3 WMMAs per tile-product, f32 accumulate) for fp32-class accuracy.
// Operand tiles are staged global->LDS with global_load_async_to_lds_b128
// (ASYNCcnt DMA), double-buffered, overlapping DMA with WMMA compute.
// 50 unrolled iterations; device-side converged flag turns post-convergence
// steps into identity copies. Deterministic fixed-order reductions.
// ---------------------------------------------------------------------------

#define TOLERANCE 1e-4f
#define MAX_ITER  50
#define ALPHA0    0.5f

static constexpr int BATCH = 512;
static constexpr int DIN_  = 3072;
static constexpr int DD    = 4096;

static constexpr int WG_M = 64;    // workgroup C-tile rows
static constexpr int WG_N = 128;   // workgroup C-tile cols (8 waves: 2x4 of 32x32)
static constexpr int NWG  = (BATCH / WG_M) * (DD / WG_N);  // 8*32 = 256

// LDS staging geometry: 32-K slab, rows padded to 80B (16 distinct banks
// across the 16 half-wave lanes reading consecutive rows).
static constexpr int PITCH   = 80;                       // bytes per 32-elem bf16 row
static constexpr int A_BYTES = WG_M * PITCH;             // 5120
static constexpr int B_BYTES = WG_N * PITCH;             // 10240
static constexpr int STAGE   = 2 * A_BYTES + 2 * B_BYTES;// 30720 (Ahi,Alo,Bhi,Blo)

typedef __attribute__((ext_vector_type(16))) __bf16 v16bf;
typedef __attribute__((ext_vector_type(8)))  __bf16 v8bf;
typedef __attribute__((ext_vector_type(8)))  float  v8f;

// ---------------- bf16 helpers (RNE) ----------------
__device__ __forceinline__ __bf16 f2bf(float f) {
  union { float f; unsigned u; } v; v.f = f;
  unsigned r = v.u + 0x7FFFu + ((v.u >> 16) & 1u);
  union { unsigned short s; __bf16 b; } o; o.s = (unsigned short)(r >> 16);
  return o.b;
}
__device__ __forceinline__ float bf2f(__bf16 b) {
  union { __bf16 b; unsigned short s; } i; i.b = b;
  union { unsigned u; float f; } o; o.u = ((unsigned)i.s) << 16;
  return o.f;
}
__device__ __forceinline__ void split2(float x, __bf16& h, __bf16& l) {
  h = f2bf(x);
  l = f2bf(x - bf2f(h));
}

// ---------------- CDNA5 async global->LDS (ASYNCcnt DMA) ----------------
// the LDS destination comes from VDST, which carries the byte offset within the
// workgroup LDS allocation (= low 32 bits of the generic shared pointer).
__device__ __forceinline__ void async_b128(unsigned lds_off, const void* gptr) {
  asm volatile("global_load_async_to_lds_b128 %0, %1, off"
               :: "v"(lds_off), "v"(gptr) : "memory");
}
__device__ __forceinline__ void wait_async_le6() {
  asm volatile("s_wait_asynccnt 0x6" ::: "memory");
}
__device__ __forceinline__ void wait_async_0() {
  asm volatile("s_wait_asynccnt 0x0" ::: "memory");
}

// ---------------- LDS fragment readers ----------------
// A-matrix 16x32 bf16 (MxK): lanes 0-15 hold M=lane, K {0..7,16..23};
// lanes 16-31 hold M=lane-16, K {8..15,24..31}.
__device__ __forceinline__ v16bf lds_frag_a(const char* base, int row0, int lane) {
  const int m  = lane & 15;
  const int ks = (lane >> 4) << 3;
  const __bf16* p = (const __bf16*)(base + (row0 + m) * PITCH) + ks;
  v8bf c0 = *(const v8bf*)p;          // K = ks .. ks+7
  v8bf c1 = *(const v8bf*)(p + 16);   // K = ks+16 .. ks+23
  v16bf f;
#pragma unroll
  for (int i = 0; i < 8; ++i) { f[i] = c0[i]; f[i + 8] = c1[i]; }
  return f;
}
// B-matrix 32x16 bf16 (KxN): lanes 0-15 hold N=lane, K {0..7,16..23}; lanes 16-31
// hold N=lane-16, K {8..15,24..31} (column n of B == row n of row-major W slab).
__device__ __forceinline__ v16bf lds_frag_b(const char* base, int row0, int lane) {
  const int n  = lane & 15;
  const int ks = (lane >> 4) << 3;
  const __bf16* p = (const __bf16*)(base + (row0 + n) * PITCH) + ks;
  v8bf c0 = *(const v8bf*)p;
  v8bf c1 = *(const v8bf*)(p + 16);
  v16bf f;
#pragma unroll
  for (int i = 0; i < 8; ++i) { f[i] = c0[i]; f[i + 8] = c1[i]; }
  return f;
}

// ---------------- utility kernels ----------------
__global__ void split_kernel(const float* __restrict__ src,
                             __bf16* __restrict__ hi, __bf16* __restrict__ lo, long n) {
  long i = (long)blockIdx.x * blockDim.x + threadIdx.x;
  const long stride = (long)gridDim.x * blockDim.x;
  for (; i < n; i += stride) {
    __bf16 h, l; split2(src[i], h, l);
    hi[i] = h; lo[i] = l;
  }
}

__global__ void init_kernel(float* __restrict__ z, __bf16* __restrict__ zh,
                            __bf16* __restrict__ zl, float* __restrict__ state, long n) {
  long i = (long)blockIdx.x * blockDim.x + threadIdx.x;
  const long stride = (long)gridDim.x * blockDim.x;
  const __bf16 bz = f2bf(0.f);
  for (; i < n; i += stride) { z[i] = 0.f; zh[i] = bz; zl[i] = bz; }
  if (blockIdx.x == 0 && threadIdx.x == 0) {
    state[0] = 1.0f;    // err
    state[1] = ALPHA0;  // a
    state[2] = 0.0f;    // converged flag
    state[3] = 0.0f;    // iteration count
  }
}

// Deterministic scalar update: err/alpha/converged, fixed-order partial sum.
__global__ void update_kernel(float* __restrict__ state,
                              const float* __restrict__ partials, int nwg) {
  if (blockIdx.x != 0 || threadIdx.x != 0) return;
  if (state[2] != 0.f) return;  // already converged: while-loop exited
  float s1 = 0.f, s2 = 0.f;
  for (int i = 0; i < nwg; ++i) { s1 += partials[2 * i]; s2 += partials[2 * i + 1]; }
  const float err_new = sqrtf(s1) / (1e-6f + sqrtf(s2));
  const float err = state[0];
  float a = state[1];
  if ((err_new > 0.85f * err) && (a > 1e-3f)) a = a / 1.5f;
  state[0] = err_new;
  state[1] = a;
  state[3] += 1.0f;
  if (!(err_new > TOLERANCE)) state[2] = 1.0f;
}

// ---------------- fused bf16x3 WMMA GEMM: C = A * B^T (+ epilogue) ----------
// A: M x K (hi/lo bf16, row-major, ld=K)   B: N x K (hi/lo bf16, row-major)
// MODE 0: outF = C + addRow[n]                       (bias = x U^T + b)
// MODE 1: zn = (1-a)*z + a*relu(C + addMat); write zn (f32 + hi/lo split),
//         accumulate ||zn-z||^2, ||zn||^2 into per-WG partials.
//         If converged flag set: identity-copy z (keeps ping-pong parity).
// MODE 2: outF = relu(C + addMat)                    (final output)
template <int MODE>
__global__ __launch_bounds__(256)
void gemm_bf16x3_kernel(const __bf16* __restrict__ Ahi, const __bf16* __restrict__ Alo,
                        const __bf16* __restrict__ Bhi, const __bf16* __restrict__ Blo,
                        int K, int N,
                        const float* __restrict__ addRow,   // MODE 0
                        const float* __restrict__ addMat,   // MODE 1/2
                        const float* __restrict__ zin,      // MODE 1
                        float* __restrict__ outF,
                        __bf16* __restrict__ outHi, __bf16* __restrict__ outLo,
                        const float* __restrict__ state,
                        float* __restrict__ partials) {
  __shared__ __align__(16) char smem[2 * STAGE];  // 60 KB double-buffered stage

  const int tid   = threadIdx.x;
  const int lane  = tid & 31;
  const int wave  = tid >> 5;
  const int waveM = wave >> 2;     // 0..1
  const int waveN = wave & 3;      // 0..3
  const int blockM = blockIdx.y * WG_M;
  const int blockN = blockIdx.x * WG_N;
  const int tileM = blockM + waveM * 32;
  const int tileN = blockN + waveN * 32;

  float a_step = 0.f;
  bool skip = false;
  if (MODE == 1) {
    a_step = state[1];
    skip = (state[2] != 0.f);
  }

  if (MODE == 1 && skip) {
    // Converged earlier: this launch is an identity step.
#pragma unroll
    for (int mi = 0; mi < 2; ++mi)
#pragma unroll
      for (int ni = 0; ni < 2; ++ni)
#pragma unroll
        for (int i = 0; i < 8; ++i) {
          const int m = tileM + mi * 16 + ((lane >> 4) << 3) + i;
          const int n = tileN + ni * 16 + (lane & 15);
          const long idx = (long)m * N + n;
          const float z = zin[idx];
          outF[idx] = z;
          __bf16 h, l; split2(z, h, l);
          outHi[idx] = h; outLo[idx] = l;
        }
    return;
  }

  // Per-thread async chunk coordinates: 16B chunks of a 32-K slab.
  // A slab: 64 rows x 4 chunks = 256 chunks -> 1/thread (x hi,lo).
  // B slab: 128 rows x 4 chunks = 512 chunks -> 2/thread (x hi,lo).
  const int arow  = tid >> 2;
  const int acb   = tid & 3;
  const int brow0 = tid >> 2;          // rows 0..63
  const int brow1 = (tid >> 2) + 64;   // rows 64..127
  const long aoff  = (long)(blockM + arow)  * K + acb * 8;
  const long b0off = (long)(blockN + brow0) * K + acb * 8;
  const long b1off = (long)(blockN + brow1) * K + acb * 8;
  const unsigned smem_base = (unsigned)(size_t)(void*)smem;  // LDS byte offset
  const unsigned aDst  = arow  * PITCH + acb * 16;
  const unsigned b0Dst = 2 * A_BYTES + brow0 * PITCH + acb * 16;
  const unsigned b1Dst = 2 * A_BYTES + brow1 * PITCH + acb * 16;

  auto issue_stage = [&](int stage, int k0) {
    const unsigned sb = smem_base + stage * STAGE;
    async_b128(sb + aDst,            Ahi + aoff  + k0);
    async_b128(sb + aDst + A_BYTES,  Alo + aoff  + k0);
    async_b128(sb + b0Dst,           Bhi + b0off + k0);
    async_b128(sb + b0Dst + B_BYTES, Blo + b0off + k0);
    async_b128(sb + b1Dst,           Bhi + b1off + k0);
    async_b128(sb + b1Dst + B_BYTES, Blo + b1off + k0);
  };

  v8f acc[2][2] = {};

  issue_stage(0, 0);
  int cur = 0;
  for (int k0 = 0; k0 < K; k0 += 32) {
    const bool more = (k0 + 32) < K;
    if (more) issue_stage(cur ^ 1, k0 + 32);  // overlap next slab DMA with compute
    if (more) wait_async_le6(); else wait_async_0();  // own slab-cur writes done
    __syncthreads();                                   // all waves' writes visible

    const char* sm = smem + cur * STAGE;
    v16bf ah[2], al[2], bh[2], bl[2];
#pragma unroll
    for (int mi = 0; mi < 2; ++mi) {
      ah[mi] = lds_frag_a(sm,           waveM * 32 + mi * 16, lane);
      al[mi] = lds_frag_a(sm + A_BYTES, waveM * 32 + mi * 16, lane);
    }
#pragma unroll
    for (int ni = 0; ni < 2; ++ni) {
      bh[ni] = lds_frag_b(sm + 2 * A_BYTES,           waveN * 32 + ni * 16, lane);
      bl[ni] = lds_frag_b(sm + 2 * A_BYTES + B_BYTES, waveN * 32 + ni * 16, lane);
    }
#pragma unroll
    for (int mi = 0; mi < 2; ++mi)
#pragma unroll
      for (int ni = 0; ni < 2; ++ni) {
        acc[mi][ni] = __builtin_amdgcn_wmma_f32_16x16x32_bf16(
            false, ah[mi], false, bh[ni], (short)0, acc[mi][ni], false, false);
        acc[mi][ni] = __builtin_amdgcn_wmma_f32_16x16x32_bf16(
            false, ah[mi], false, bl[ni], (short)0, acc[mi][ni], false, false);
        acc[mi][ni] = __builtin_amdgcn_wmma_f32_16x16x32_bf16(
            false, al[mi], false, bh[ni], (short)0, acc[mi][ni], false, false);
      }

    __syncthreads();  // all reads of slab-cur done before it is overwritten
    cur ^= 1;
  }

  float s1 = 0.f, s2 = 0.f;
  // C/D layout: VGPR i, lanes 0-15 -> (M=i, N=lane); lanes 16-31 -> (M=8+i, N=lane-16)
#pragma unroll
  for (int mi = 0; mi < 2; ++mi)
#pragma unroll
    for (int ni = 0; ni < 2; ++ni)
#pragma unroll
      for (int i = 0; i < 8; ++i) {
        const int m = tileM + mi * 16 + ((lane >> 4) << 3) + i;
        const int n = tileN + ni * 16 + (lane & 15);
        const long idx = (long)m * N + n;
        const float v = acc[mi][ni][i];
        if (MODE == 0) {
          outF[idx] = v + addRow[n];
        } else if (MODE == 2) {
          const float pre = v + addMat[idx];
          outF[idx] = pre > 0.f ? pre : 0.f;
        } else {
          const float pre = v + addMat[idx];
          const float r = pre > 0.f ? pre : 0.f;
          const float z = zin[idx];
          const float zn = (1.f - a_step) * z + a_step * r;
          outF[idx] = zn;
          __bf16 h, l; split2(zn, h, l);
          outHi[idx] = h; outLo[idx] = l;
          const float d = zn - z;
          s1 += d * d;
          s2 += zn * zn;
        }
      }

  if (MODE == 1) {
    // Reuse the (fully consumed) staging LDS for the deterministic reduction.
    float* red1 = (float*)smem;
    float* red2 = red1 + 256;
    red1[tid] = s1;
    red2[tid] = s2;
    __syncthreads();
    for (int s = 128; s > 0; s >>= 1) {
      if (tid < s) {
        red1[tid] += red1[tid + s];
        red2[tid] += red2[tid + s];
      }
      __syncthreads();
    }
    if (tid == 0) {
      const int wg = blockIdx.y * gridDim.x + blockIdx.x;
      partials[2 * wg]     = red1[0];
      partials[2 * wg + 1] = red2[0];
    }
  }
}

// ---------------------------------------------------------------------------
extern "C" void kernel_launch(void* const* d_in, const int* in_sizes, int n_in,
                              void* d_out, int out_size, void* d_ws, size_t ws_size,
                              hipStream_t stream) {
  (void)in_sizes; (void)n_in; (void)out_size; (void)ws_size;
  const float* x = (const float*)d_in[0];  // (B, DIN)
  const float* W = (const float*)d_in[1];  // (D, D)
  const float* U = (const float*)d_in[2];  // (D, DIN)
  const float* b = (const float*)d_in[3];  // (D,)

  char* ws = (char*)d_ws;
  size_t off = 0;
  auto carve = [&](size_t bytes) -> char* {
    off = (off + 255) & ~(size_t)255;
    char* p = ws + off;
    off += bytes;
    return p;
  };

  __bf16* Whi = (__bf16*)carve((size_t)DD * DD * 2);
  __bf16* Wlo = (__bf16*)carve((size_t)DD * DD * 2);
  __bf16* Uhi = (__bf16*)carve((size_t)DD * DIN_ * 2);
  __bf16* Ulo = (__bf16*)carve((size_t)DD * DIN_ * 2);
  __bf16* xhi = (__bf16*)carve((size_t)BATCH * DIN_ * 2);
  __bf16* xlo = (__bf16*)carve((size_t)BATCH * DIN_ * 2);
  float*  biasbuf = (float*)carve((size_t)BATCH * DD * 4);
  float*  z0  = (float*)carve((size_t)BATCH * DD * 4);
  float*  z1  = (float*)carve((size_t)BATCH * DD * 4);
  __bf16* z0h = (__bf16*)carve((size_t)BATCH * DD * 2);
  __bf16* z0l = (__bf16*)carve((size_t)BATCH * DD * 2);
  __bf16* z1h = (__bf16*)carve((size_t)BATCH * DD * 2);
  __bf16* z1l = (__bf16*)carve((size_t)BATCH * DD * 2);
  float*  partials = (float*)carve((size_t)2 * NWG * 4);
  float*  state    = (float*)carve(4 * 4);

  // 1) hi/lo bf16 splits of the constant operands (amortized over 51 GEMMs)
  split_kernel<<<2048, 256, 0, stream>>>(W, Whi, Wlo, (long)DD * DD);
  split_kernel<<<2048, 256, 0, stream>>>(U, Uhi, Ulo, (long)DD * DIN_);
  split_kernel<<<512, 256, 0, stream>>>(x, xhi, xlo, (long)BATCH * DIN_);

  // 2) z0 = 0, state = {err=1, a=0.5, conv=0, it=0}
  init_kernel<<<2048, 256, 0, stream>>>(z0, z0h, z0l, state, (long)BATCH * DD);

  const dim3 gG(DD / WG_N, BATCH / WG_M);  // (32, 8) = 256 workgroups

  // 3) bias = x @ U^T + b
  gemm_bf16x3_kernel<0><<<gG, 256, 0, stream>>>(
      xhi, xlo, Uhi, Ulo, DIN_, DD, b, nullptr, nullptr,
      biasbuf, nullptr, nullptr, state, partials);

  // 4) 50 unrolled averaged fixed-point steps (flag-gated once converged)
  for (int it = 0; it < MAX_ITER; ++it) {
    const bool even = (it & 1) == 0;
    const float*  zi  = even ? z0  : z1;
    float*        zo  = even ? z1  : z0;
    const __bf16* zih = even ? z0h : z1h;
    const __bf16* zil = even ? z0l : z1l;
    __bf16*       zoh = even ? z1h : z0h;
    __bf16*       zol = even ? z1l : z0l;
    gemm_bf16x3_kernel<1><<<gG, 256, 0, stream>>>(
        zih, zil, Whi, Wlo, DD, DD, nullptr, biasbuf, zi,
        zo, zoh, zol, state, partials);
    update_kernel<<<1, 32, 0, stream>>>(state, partials, NWG);
  }

  // 5) out = relu(z @ W^T + bias); after 50 ping-pong steps z lives in z0
  gemm_bf16x3_kernel<2><<<gG, 256, 0, stream>>>(
      z0h, z0l, Whi, Wlo, DD, DD, nullptr, biasbuf, nullptr,
      (float*)d_out, nullptr, nullptr, state, partials);
}
